// JointGraphAttention_66786741453300
// MI455X (gfx1250) — matrix operand order repaired and checked by reading, not compiled, not run on hardware
//
#include <hip/hip_runtime.h>
#include <hip/hip_bf16.h>

typedef _Float16 v8h  __attribute__((ext_vector_type(8)));
typedef _Float16 v16h __attribute__((ext_vector_type(16)));
typedef float    v8f  __attribute__((ext_vector_type(8)));

#define IDX16 0,1,2,3,4,5,6,7,8,9,10,11,12,13,14,15

// ---------------------------------------------------------------------------
// CDNA5 async-copy helpers (ASYNCcnt-tracked global->LDS DMA, no VGPR staging)
// ---------------------------------------------------------------------------
__device__ __forceinline__ unsigned lds_addr(const void* p) {
    // flat LDS address: low 32 bits are the wave-relative LDS byte offset
    return (unsigned)(unsigned long long)p;
}
__device__ __forceinline__ void async_b128(unsigned lds, const void* g) {
    asm volatile("global_load_async_to_lds_b128 %0, %1, off"
                 :: "v"(lds), "v"((unsigned long long)g) : "memory");
}
__device__ __forceinline__ void async_b64(unsigned lds, const void* g) {
    asm volatile("global_load_async_to_lds_b64 %0, %1, off"
                 :: "v"(lds), "v"((unsigned long long)g) : "memory");
}
template <int N>
__device__ __forceinline__ void wait_asynccnt() {
    asm volatile("s_wait_asynccnt %0" :: "i"(N) : "memory");
}

// ---------------------------------------------------------------------------
// Elementwise prep kernels
// ---------------------------------------------------------------------------

// W fp32 [in,out] -> Wt f16 [out,in]
__global__ __launch_bounds__(256) void wtrans_kernel(const float* __restrict__ W,
                                                     _Float16* __restrict__ Wt,
                                                     int in, int out) {
    int i = blockIdx.x * 256 + threadIdx.x;
    if (i >= in * out) return;
    int o  = i / in;
    int ii = i - o * in;
    Wt[i] = (_Float16)W[(long)ii * out + o];
}

// fp32 -> f16 elementwise
__global__ __launch_bounds__(256) void cvt16_kernel(const float* __restrict__ src,
                                                    _Float16* __restrict__ dst, int n) {
    int i = blockIdx.x * 256 + threadIdx.x;
    if (i < n) dst[i] = (_Float16)src[i];
}

// Timestep embedding: query_pos [4096] -> emb f16 [4096, 256] (cos | sin)
__global__ __launch_bounds__(256) void embed_kernel(const float* __restrict__ qp,
                                                    _Float16* __restrict__ emb) {
    int i = blockIdx.x * 256 + threadIdx.x;     // over 4096*128
    if (i >= 4096 * 128) return;
    int row = i >> 7;
    int j   = i & 127;
    float f  = __expf(-9.210340371976184f * (float)j * (1.0f / 128.0f));
    float tf = qp[row] * f;
    emb[(long)row * 256 + j]       = (_Float16)__cosf(tf);
    emb[(long)row * 256 + 128 + j] = (_Float16)__sinf(tf);
}

// V fp32 [b*64+m, h*64+d] -> Vt f16 [((b*8+h)*64+d)*64 + m]
__global__ __launch_bounds__(256) void vtrans_kernel(const float* __restrict__ V32,
                                                     _Float16* __restrict__ Vt) {
    int i = blockIdx.x * 256 + threadIdx.x;     // over 4096*512
    if (i >= 4096 * 512) return;
    int row = i >> 9, col = i & 511;
    int b = row >> 6, m = row & 63;
    int h = col >> 6, d = col & 63;
    Vt[(((long)(b * 8 + h) * 64 + d) * 64) + m] = (_Float16)V32[i];
}

// ---------------------------------------------------------------------------
// WMMA GEMM with async-LDS double-buffered staging:
//   out[rows, ncols] = A[rows,K] @ Bt[ncols,K]^T (+bias)(+epi)
// block = 4 waves = 16 rows x 256 cols; per K-step tile: A 16x32, B 256x32.
// Each thread issues 9 async ops/stage (1 b64 A + 8 b128 B) -> uniform
// s_wait_asynccnt 9 retires the previous stage while the next is in flight.
// EPI: 1 = silu -> f16 ; 2 = f32 ; 3 = +residual -> f32
// ---------------------------------------------------------------------------
template <int EPI>
__global__ __launch_bounds__(128) void gemm_wmma_kernel(
    const _Float16* __restrict__ A, const _Float16* __restrict__ Bt,
    const float* __restrict__ bias, const float* __restrict__ resid,
    void* __restrict__ outp, int K, int ncols) {
    __shared__ _Float16 As[2][16 * 32];     //  2 KB
    __shared__ _Float16 Bs[2][256 * 32];    // 32 KB
    const int tid  = threadIdx.x;
    const int lane = tid & 31;
    const int wid  = tid >> 5;
    const int l15  = lane & 15;
    const int hi   = lane >> 4;
    const long rowBase  = (long)blockIdx.x * 16;
    const int  colBase0 = blockIdx.y * 256;   // block's 256-col strip
    const int  waveCol  = wid * 64;           // wave's 64 cols within strip

    auto stage = [&](int k0, int buf) {
        // A tile: 16 rows x 32 k (1 KB) = 128 threads x 1 b64
        {
            int row = tid >> 2, kc = tid & 3;          // kc: 4-f16 (8B) chunk
            async_b64(lds_addr(&As[buf][row * 32 + kc * 4]),
                      A + (rowBase + row) * K + k0 + kc * 4);
        }
        // B tile: 256 cols x 32 k (16 KB) = 128 threads x 8 b128
#pragma unroll
        for (int j = 0; j < 8; ++j) {
            int c   = tid + j * 128;                   // 0..1023 16B chunks
            int col = c >> 2, kc = c & 3;              // kc: 8-f16 (16B) chunk
            async_b128(lds_addr(&Bs[buf][col * 32 + kc * 8]),
                       Bt + (long)(colBase0 + col) * K + k0 + kc * 8);
        }
    };

    v8f acc0 = {}, acc1 = {}, acc2 = {}, acc3 = {};
    const int S = K / 32;
    stage(0, 0);
    for (int s = 0; s < S; ++s) {
        if (s + 1 < S) {
            stage((s + 1) * 32, (s + 1) & 1);
            wait_asynccnt<9>();     // previous stage's 9 ops retired
        } else {
            wait_asynccnt<0>();
        }
        __syncthreads();
        const int buf = s & 1;
        // A fragment: lane-lo K {0..15}; lane-hi K {16..31}
        v8h a0 = *(const v8h*)&As[buf][l15 * 32 + hi * 16];
        v8h a1 = *(const v8h*)&As[buf][l15 * 32 + hi * 16 + 8];
        v16h a = __builtin_shufflevector(a0, a1, IDX16);
        v8f* accs[4] = {&acc0, &acc1, &acc2, &acc3};
#pragma unroll
        for (int j = 0; j < 4; ++j) {
            int colL = waveCol + j * 16 + l15;         // B lane = column
            v8h b0 = *(const v8h*)&Bs[buf][colL * 32 + hi * 16];
            v8h b1 = *(const v8h*)&Bs[buf][colL * 32 + hi * 16 + 8];
            v16h b = __builtin_shufflevector(b0, b1, IDX16);
            *accs[j] = __builtin_amdgcn_wmma_f32_16x16x32_f16(
                false, a, false, b, (short)0, *accs[j], false, false);
        }
        __syncthreads();   // all waves done with buf before it is re-staged
    }

    v8f accs[4] = {acc0, acc1, acc2, acc3};
#pragma unroll
    for (int j = 0; j < 4; ++j) {
        int col  = colBase0 + waveCol + j * 16 + l15;
        float bv = bias ? bias[col] : 0.0f;
#pragma unroll
        for (int r = 0; r < 8; ++r) {
            long row = rowBase + r + hi * 8;   // C/D layout: VGPR r -> M=r (+8 hi)
            float v  = accs[j][r] + bv;
            long idx = row * ncols + col;
            if (EPI == 1) {
                v = v / (1.0f + __expf(-v));   // SiLU
                ((_Float16*)outp)[idx] = (_Float16)v;
            } else if (EPI == 2) {
                ((float*)outp)[idx] = v;
            } else {
                ((float*)outp)[idx] = v + resid[idx];
            }
        }
    }
}

// ---------------------------------------------------------------------------
// Attention: one block per (b,h). Async-LDS q/k staging, trilinear scores
// (VALU), softmax, then attn@V as WMMA fed straight from LDS.
// ---------------------------------------------------------------------------
__global__ __launch_bounds__(256) void attn_kernel(
    const float* __restrict__ Q, const float* __restrict__ Kb,
    const float* __restrict__ PE, const _Float16* __restrict__ Vt,
    _Float16* __restrict__ X) {
    const int bh = blockIdx.x;
    const int b  = bh >> 3;
    const int h  = bh & 7;
    __shared__ float    qs[64 * 64];
    __shared__ float    ks[64 * 64];
    __shared__ float    sc[64 * 64];
    __shared__ _Float16 attnS[64 * 64];
    const int t = threadIdx.x;

    // stage q/k slices (16 KB each) via async DMA: 256 threads x 4 b128 each
#pragma unroll
    for (int j = 0; j < 4; ++j) {
        int c  = t + j * 256;          // 0..1023 16B chunks
        int n  = c >> 4, cc = c & 15;
        long g = (long)(b * 64 + n) * 512 + h * 64 + cc * 4;
        async_b128(lds_addr(&qs[c * 4]), Q + g);
        async_b128(lds_addr(&ks[c * 4]), Kb + g);
    }
    wait_asynccnt<0>();
    __syncthreads();

    // scores[n,m] = scale * sum_d q[n,d] * pe[n,m,d] * k[m,d]
    for (int i = t; i < 4096; i += 256) {
        int n = i >> 6, m = i & 63;
        const float* pe = PE + ((long)(n * 64 + m)) * 512 + h * 64;
        float acc = 0.0f;
#pragma unroll 8
        for (int d = 0; d < 64; ++d)
            acc += qs[n * 64 + d] * ks[m * 64 + d] * pe[d];
        sc[i] = acc * 0.125f;   // HD^-0.5
    }
    __syncthreads();

    // softmax rows (one thread per row)
    if (t < 64) {
        int n = t;
        float mx = -3.0e38f;
        for (int m = 0; m < 64; ++m) mx = fmaxf(mx, sc[n * 64 + m]);
        float s = 0.0f;
        for (int m = 0; m < 64; ++m) s += __expf(sc[n * 64 + m] - mx);
        float inv = 1.0f / s;
        for (int m = 0; m < 64; ++m)
            attnS[n * 64 + m] = (_Float16)(__expf(sc[n * 64 + m] - mx) * inv);
    }
    __syncthreads();

    // x[n,d] = sum_m attn[n,m] * v[m,d]  — 64x64x64 GEMM via WMMA
    const int lane = t & 31, wid = t >> 5, l15 = lane & 15, hi = lane >> 4;
#pragma unroll
    for (int tt = 0; tt < 2; ++tt) {
        int tile = wid * 2 + tt;
        int tr = tile >> 2, tc = tile & 3;
        v8f acc = {};
#pragma unroll
        for (int k0 = 0; k0 < 64; k0 += 32) {
            const _Float16* arow = attnS + (tr * 16 + l15) * 64 + k0 + hi * 16;
            v8h a0 = *(const v8h*)(arow);
            v8h a1 = *(const v8h*)(arow + 8);
            v16h a = __builtin_shufflevector(a0, a1, IDX16);
            int dcol = tc * 16 + l15;
            const _Float16* brow = Vt + ((long)(b * 8 + h) * 64 + dcol) * 64 + k0 + hi * 16;
            v8h b0 = *(const v8h*)(brow);
            v8h b1 = *(const v8h*)(brow + 8);
            v16h bb = __builtin_shufflevector(b0, b1, IDX16);
            acc = __builtin_amdgcn_wmma_f32_16x16x32_f16(
                false, a, false, bb, (short)0, acc, false, false);
        }
#pragma unroll
        for (int r = 0; r < 8; ++r) {
            int n   = tr * 16 + r + hi * 8;
            int col = h * 64 + tc * 16 + l15;
            X[(long)(b * 64 + n) * 512 + col] = (_Float16)acc[r];
        }
    }
}

// ---------------------------------------------------------------------------
// Host launcher
// ---------------------------------------------------------------------------
extern "C" void kernel_launch(void* const* d_in, const int* in_sizes, int n_in,
                              void* d_out, int out_size, void* d_ws, size_t ws_size,
                              hipStream_t stream) {
    (void)in_sizes; (void)n_in; (void)out_size; (void)ws_size;
    const float* query = (const float*)d_in[0];   // [64*64, 512]
    const float* qpos  = (const float*)d_in[1];   // [4096]
    const float* Wq    = (const float*)d_in[2];
    const float* bq    = (const float*)d_in[3];
    const float* Wk    = (const float*)d_in[4];
    const float* Wv    = (const float*)d_in[5];
    const float* bv    = (const float*)d_in[6];
    const float* Wo    = (const float*)d_in[7];
    const float* bo    = (const float*)d_in[8];
    const float* W1    = (const float*)d_in[9];
    const float* b1    = (const float*)d_in[10];
    const float* W2    = (const float*)d_in[11];
    const float* b2    = (const float*)d_in[12];
    float* out = (float*)d_out;

    char* ws = (char*)d_ws;
    size_t off = 0;
    auto alloc = [&](size_t bytes) { char* p = ws + off; off += (bytes + 255) & ~255ULL; return p; };
    _Float16* W1t = (_Float16*)alloc(512 * 256 * 2);
    _Float16* W2t = (_Float16*)alloc(512 * 512 * 2);
    _Float16* Wqt = (_Float16*)alloc(512 * 512 * 2);
    _Float16* Wkt = (_Float16*)alloc(512 * 512 * 2);
    _Float16* Wvt = (_Float16*)alloc(512 * 512 * 2);
    _Float16* Wot = (_Float16*)alloc(512 * 512 * 2);
    _Float16* qA  = (_Float16*)alloc((size_t)4096 * 512 * 2);
    _Float16* emb = (_Float16*)alloc((size_t)4096 * 256 * 2);
    _Float16* H1  = (_Float16*)alloc((size_t)4096 * 512 * 2);
    float*    PE  = (float*)   alloc((size_t)4096 * 512 * 4);
    float*    Qb  = (float*)   alloc((size_t)4096 * 512 * 4);
    float*    Kb  = (float*)   alloc((size_t)4096 * 512 * 4);
    float*    V32 = (float*)   alloc((size_t)4096 * 512 * 4);
    _Float16* Vt  = (_Float16*)alloc((size_t)4096 * 512 * 2);
    _Float16* X   = (_Float16*)alloc((size_t)4096 * 512 * 2);

    // --- weight prep (fp32 [in,out] -> f16 [out,in]) ---
    wtrans_kernel<<<(256 * 512 + 255) / 256, 256, 0, stream>>>(W1, W1t, 256, 512);
    wtrans_kernel<<<(512 * 512 + 255) / 256, 256, 0, stream>>>(W2, W2t, 512, 512);
    wtrans_kernel<<<(512 * 512 + 255) / 256, 256, 0, stream>>>(Wq, Wqt, 512, 512);
    wtrans_kernel<<<(512 * 512 + 255) / 256, 256, 0, stream>>>(Wk, Wkt, 512, 512);
    wtrans_kernel<<<(512 * 512 + 255) / 256, 256, 0, stream>>>(Wv, Wvt, 512, 512);
    wtrans_kernel<<<(512 * 512 + 255) / 256, 256, 0, stream>>>(Wo, Wot, 512, 512);
    cvt16_kernel<<<(4096 * 512 + 255) / 256, 256, 0, stream>>>(query, qA, 4096 * 512);
    embed_kernel<<<(4096 * 128 + 255) / 256, 256, 0, stream>>>(qpos, emb);

    dim3 gemmGrid(4096 / 16, 512 / 256);
    // H1 = silu(emb @ W1 + b1) -> f16
    gemm_wmma_kernel<1><<<gemmGrid, 128, 0, stream>>>(emb, W1t, b1, nullptr, H1, 256, 512);
    // PE = H1 @ W2 + b2 -> f32
    gemm_wmma_kernel<2><<<gemmGrid, 128, 0, stream>>>(H1, W2t, b2, nullptr, PE, 512, 512);
    // Q, K, V projections
    gemm_wmma_kernel<2><<<gemmGrid, 128, 0, stream>>>(qA, Wqt, bq, nullptr, Qb, 512, 512);
    gemm_wmma_kernel<2><<<gemmGrid, 128, 0, stream>>>(qA, Wkt, nullptr, nullptr, Kb, 512, 512);
    gemm_wmma_kernel<2><<<gemmGrid, 128, 0, stream>>>(qA, Wvt, bv, nullptr, V32, 512, 512);
    vtrans_kernel<<<(4096 * 512 + 255) / 256, 256, 0, stream>>>(V32, Vt);

    // attention per (b,h)
    attn_kernel<<<64 * 8, 256, 0, stream>>>(Qb, Kb, PE, Vt, X);

    // out = X @ Wo + bo + query -> f32
    gemm_wmma_kernel<3><<<gemmGrid, 128, 0, stream>>>(X, Wot, bo, query, out, 512, 512);
}
